// TripletTransformerNetwork_31009663877394
// MI455X (gfx1250) — compile-verified, matrix operand order checked by reading, not run on hardware
//
#include <hip/hip_runtime.h>
#include <hip/hip_bf16.h>

#define N_NODES 2048
#define DIM     512
#define HEADS   8
#define HDIM    (HEADS * DIM)   // 4096
#define LAYERS  6
#define NCLASS  10
#define LN_EPSF 1e-5f

typedef __bf16 bf16;
typedef __attribute__((ext_vector_type(16))) __bf16 bf16x16;
typedef __attribute__((ext_vector_type(8)))  __bf16 bf16x8;
typedef __attribute__((ext_vector_type(8)))  float  f32x8;
typedef __attribute__((ext_vector_type(4)))  int    i32x4;

enum { ST_F32 = 1, ST_BF16 = 2, ST_BF16_T = 4, EPI_SCORES = 8 };

#define BM 128
#define BN 128
#define BK 32
#define LSTR 40   // padded LDS row stride (elements), keeps 16B alignment

// ---- CDNA5 async global->LDS copy path (probe at compile time) ------------
#if defined(__AMDGCN__) && __has_builtin(__builtin_amdgcn_global_load_async_to_lds_b128)
#define USE_ASYNC_LDS 1
#define GAS1(p) ((__attribute__((address_space(1))) i32x4*)(p))
#define LAS3(p) ((__attribute__((address_space(3))) i32x4*)(p))
#else
#define USE_ASYNC_LDS 0
#endif

#if defined(__AMDGCN__) && __has_builtin(__builtin_amdgcn_s_wait_asynccnt)
#define WAIT_ASYNCCNT(n) __builtin_amdgcn_s_wait_asynccnt(n)
#elif USE_ASYNC_LDS
#define WAIT_ASYNCCNT(n) asm volatile("s_wait_asynccnt %0" ::"i"(n) : "memory")
#else
#define WAIT_ASYNCCNT(n) ((void)0)
#endif

// ---------------------------------------------------------------------------
// Batched bf16 WMMA GEMM: C = alpha * A(MxK) * Bt(NxK)^T (+bias) (+score bias)
// A row-major [M x K], Bt row-major [N x K] (i.e. B pre-transposed).
// blockDim = 256 (8 wave32). Block tile 128x128, K-step 32, double-buffered
// LDS stages filled by GLOBAL_LOAD_ASYNC_TO_LDS_B128 (ASYNCcnt pipeline).
// ---------------------------------------------------------------------------
template <int FLAGS>
__global__ __launch_bounds__(256) void gemm_bf16_kernel(
    const bf16* __restrict__ A, long lda, long strideA,
    const bf16* __restrict__ Bt, long ldb, long strideB,
    const float* __restrict__ bias,
    float* __restrict__ Cf, bf16* __restrict__ Cb,
    long ldc, long strideC,
    int M, int Nn, int Kk, float alpha,
    const float* __restrict__ locs, const float* __restrict__ mu,
    const float* __restrict__ sigma, const float* __restrict__ Wg,
    const float* __restrict__ bg)
{
    __shared__ __align__(16) bf16 lA[2][BM][LSTR];   // 2 x 10 KB
    __shared__ __align__(16) bf16 lB[2][BN][LSTR];   // 2 x 10 KB

    const int tid   = threadIdx.x;
    const int lane  = tid & 31;
    const int wid   = tid >> 5;
    const int waveM = wid >> 1;          // 0..3 -> 32-row slab
    const int waveN = wid & 1;           // 0..1 -> 64-col slab
    const int rA    = lane & 15;
    const int kb    = (lane >> 4) * 8;   // K base per 16-bit fragment layout

    const bf16* gA = A  + (size_t)blockIdx.z * strideA + (size_t)blockIdx.y * BM * lda;
    const bf16* gB = Bt + (size_t)blockIdx.z * strideB + (size_t)blockIdx.x * BN * ldb;

    // stage copy: 128x32 bf16 tile of A and of B -> LDS buffer `buf`
    auto stage = [&](int buf, int kt) {
#pragma unroll
        for (int i = 0; i < 2; ++i) {
            int idx = tid + i * 256;
            int row = idx >> 2;
            int cc  = (idx & 3) * 8;
#if USE_ASYNC_LDS
            __builtin_amdgcn_global_load_async_to_lds_b128(
                GAS1(gA + (size_t)row * lda + kt + cc),
                LAS3(&lA[buf][row][cc]), 0, 0);
            __builtin_amdgcn_global_load_async_to_lds_b128(
                GAS1(gB + (size_t)row * ldb + kt + cc),
                LAS3(&lB[buf][row][cc]), 0, 0);
#else
            *(bf16x8*)&lA[buf][row][cc] = *(const bf16x8*)&gA[(size_t)row * lda + kt + cc];
            *(bf16x8*)&lB[buf][row][cc] = *(const bf16x8*)&gB[(size_t)row * ldb + kt + cc];
#endif
        }
    };

    f32x8 zero;
#pragma unroll
    for (int e = 0; e < 8; ++e) zero[e] = 0.f;
    f32x8 acc[2][4];
#pragma unroll
    for (int i = 0; i < 2; ++i)
#pragma unroll
        for (int j = 0; j < 4; ++j) acc[i][j] = zero;

    stage(0, 0);                 // prologue: fill stage 0 (4 async ops in flight)
    int b = 0;

    for (int kt = 0; kt < Kk; kt += BK) {
        if (kt + BK < Kk) {
            stage(b ^ 1, kt + BK);   // overlap next tile's DMA with this compute
            WAIT_ASYNCCNT(4);        // retire current stage's 4 copies
        } else {
            WAIT_ASYNCCNT(0);
        }
        __syncthreads();             // all waves' current stage resident

        bf16x16 af[2], bfg[4];
#pragma unroll
        for (int tm = 0; tm < 2; ++tm) {
            int r = waveM * 32 + tm * 16 + rA;
            union { bf16x16 v; bf16x8 h[2]; } u;
            u.h[0] = *(const bf16x8*)&lA[b][r][kb];
            u.h[1] = *(const bf16x8*)&lA[b][r][kb + 16];
            af[tm] = u.v;
        }
#pragma unroll
        for (int tn = 0; tn < 4; ++tn) {
            int c = waveN * 64 + tn * 16 + rA;
            union { bf16x16 v; bf16x8 h[2]; } u;
            u.h[0] = *(const bf16x8*)&lB[b][c][kb];
            u.h[1] = *(const bf16x8*)&lB[b][c][kb + 16];
            bfg[tn] = u.v;
        }
#pragma unroll
        for (int tm = 0; tm < 2; ++tm)
#pragma unroll
            for (int tn = 0; tn < 4; ++tn)
                acc[tm][tn] = __builtin_amdgcn_wmma_f32_16x16x32_bf16(
                    false, af[tm], false, bfg[tn], (short)0, acc[tm][tn],
                    false, false);

        __syncthreads();             // reads done before next DMA overwrites buf
        b ^= 1;
    }

    // ---------------- epilogue ----------------
    const long   zC      = (long)blockIdx.z * strideC;
    const size_t rowBase = (size_t)blockIdx.y * BM + waveM * 32 + ((lane >> 4) << 3);
    const int    colBase = blockIdx.x * BN + waveN * 64 + (lane & 15);

    float m0 = 0.f, m1 = 0.f, iv0 = 0.f, iv1 = 0.f, w0 = 0.f, w1 = 0.f, bb = 0.f;
    if (FLAGS & EPI_SCORES) {
        int h = blockIdx.z;
        m0 = mu[0]; m1 = mu[1];
        float s0 = sigma[0], s1 = sigma[1];
        iv0 = -0.5f / (s0 * s0 + 1e-6f);
        iv1 = -0.5f / (s1 * s1 + 1e-6f);
        w0 = Wg[h]; w1 = Wg[HEADS + h]; bb = bg[h];
    }

#pragma unroll
    for (int tm = 0; tm < 2; ++tm) {
#pragma unroll
        for (int tn = 0; tn < 4; ++tn) {
            int   col  = colBase + tn * 16;
            float bcol = (bias != nullptr) ? bias[col] : 0.f;
            float jx = 0.f, jy = 0.f;
            if (FLAGS & EPI_SCORES) { jx = locs[2 * col]; jy = locs[2 * col + 1]; }
#pragma unroll
            for (int r = 0; r < 8; ++r) {
                size_t row = rowBase + tm * 16 + r;
                float  v   = acc[tm][tn][r] * alpha + bcol;
                if (FLAGS & EPI_SCORES) {
                    float dx = locs[2 * row] - jx;
                    float dy = locs[2 * row + 1] - jy;
                    float dist = sqrtf(dx * dx + dy * dy);
                    float d0 = dist - m0, d1 = dist - m1;
                    float gb = __expf(d0 * d0 * iv0) * w0 +
                               __expf(d1 * d1 * iv1) * w1 + bb;
                    v += fmaxf(gb, 0.f);
                }
                if (FLAGS & ST_F32)    Cf[zC + row * ldc + col] = v;
                if (FLAGS & ST_BF16)   Cb[zC + row * ldc + col] = (bf16)v;
                if (FLAGS & ST_BF16_T) Cb[zC + (size_t)col * ldc + row] = (bf16)v;
            }
        }
    }
}

// ---------------------------------------------------------------------------
// f32 -> bf16 elementwise
// ---------------------------------------------------------------------------
__global__ void cvt_bf16_kernel(const float* __restrict__ s, bf16* __restrict__ d, int n)
{
    int i = blockIdx.x * 256 + threadIdx.x;
    if (i < n) d[i] = (bf16)s[i];
}

// f32 [rows x cols] -> bf16 transposed [cols x rows], LDS tiled
__global__ void transpose_bf16_kernel(const float* __restrict__ s, bf16* __restrict__ d,
                                      int rows, int cols)
{
    __shared__ bf16 t[32][33];
    int c0 = blockIdx.x * 32, r0 = blockIdx.y * 32;
    int tx = threadIdx.x, ty = threadIdx.y;      // block (32, 8)
#pragma unroll
    for (int i = 0; i < 4; ++i)
        t[ty + i * 8][tx] = (bf16)s[(size_t)(r0 + ty + i * 8) * cols + c0 + tx];
    __syncthreads();
#pragma unroll
    for (int i = 0; i < 4; ++i)
        d[(size_t)(c0 + ty + i * 8) * rows + r0 + tx] = t[tx][ty + i * 8];
}

// ---------------------------------------------------------------------------
// LayerNorm (+optional residual) + ELU, writes f32 and bf16. One block per row.
// ---------------------------------------------------------------------------
__global__ __launch_bounds__(256) void ln_elu_kernel(
    const float* __restrict__ t, const float* __restrict__ res,
    const float* __restrict__ g, const float* __restrict__ b,
    float* __restrict__ zf, bf16* __restrict__ zb)
{
    __shared__ float red[256];
    int    tid  = threadIdx.x;
    size_t base = (size_t)blockIdx.x * DIM;
    float x0 = t[base + tid];
    float x1 = t[base + tid + 256];
    if (res) { x0 += res[base + tid]; x1 += res[base + tid + 256]; }
    red[tid] = x0 + x1; __syncthreads();
    for (int o = 128; o > 0; o >>= 1) { if (tid < o) red[tid] += red[tid + o]; __syncthreads(); }
    float mean = red[0] * (1.f / DIM);
    __syncthreads();
    float d0 = x0 - mean, d1 = x1 - mean;
    red[tid] = d0 * d0 + d1 * d1; __syncthreads();
    for (int o = 128; o > 0; o >>= 1) { if (tid < o) red[tid] += red[tid + o]; __syncthreads(); }
    float rstd = rsqrtf(red[0] * (1.f / DIM) + LN_EPSF);
    float y0 = d0 * rstd * g[tid] + b[tid];
    float y1 = d1 * rstd * g[tid + 256] + b[tid + 256];
    y0 = y0 > 0.f ? y0 : expm1f(y0);
    y1 = y1 > 0.f ? y1 : expm1f(y1);
    zf[base + tid]       = y0;
    zf[base + tid + 256] = y1;
    zb[base + tid]       = (bf16)y0;
    zb[base + tid + 256] = (bf16)y1;
}

// ---------------------------------------------------------------------------
// Row softmax over 2048 elems, in-place f32 + bf16 copy. grid(N, H), block 256.
// ---------------------------------------------------------------------------
__global__ __launch_bounds__(256) void softmax_kernel(float* __restrict__ a,
                                                      bf16* __restrict__ ab)
{
    __shared__ float red[256];
    int    tid  = threadIdx.x;
    size_t base = ((size_t)blockIdx.y * N_NODES + blockIdx.x) * N_NODES;
    float v[8];
    float mx = -3.4e38f;
#pragma unroll
    for (int k = 0; k < 8; ++k) { v[k] = a[base + tid + k * 256]; mx = fmaxf(mx, v[k]); }
    red[tid] = mx; __syncthreads();
    for (int o = 128; o > 0; o >>= 1) { if (tid < o) red[tid] = fmaxf(red[tid], red[tid + o]); __syncthreads(); }
    mx = red[0]; __syncthreads();
    float s = 0.f;
#pragma unroll
    for (int k = 0; k < 8; ++k) { v[k] = __expf(v[k] - mx); s += v[k]; }
    red[tid] = s; __syncthreads();
    for (int o = 128; o > 0; o >>= 1) { if (tid < o) red[tid] += red[tid + o]; __syncthreads(); }
    float inv = 1.f / red[0];
#pragma unroll
    for (int k = 0; k < 8; ++k) {
        float o_ = v[k] * inv;
        a[base + tid + k * 256]  = o_;
        ab[base + tid + k * 256] = (bf16)o_;
    }
}

// ---------------------------------------------------------------------------
// Final classifier: out[n,c] = z[n,:] . W[:,c] + b[c]   (C=10 -> scalar path)
// ---------------------------------------------------------------------------
__global__ void fc1_kernel(const float* __restrict__ z, const float* __restrict__ W,
                           const float* __restrict__ b, float* __restrict__ out)
{
    int i = blockIdx.x * 256 + threadIdx.x;
    if (i >= N_NODES * NCLASS) return;
    int n = i / NCLASS, c = i % NCLASS;
    float s = b[c];
    for (int d = 0; d < DIM; ++d)
        s += z[(size_t)n * DIM + d] * W[(size_t)d * NCLASS + c];
    out[i] = s;
}

// ---------------------------------------------------------------------------
extern "C" void kernel_launch(void* const* d_in, const int* in_sizes, int n_in,
                              void* d_out, int out_size, void* d_ws, size_t ws_size,
                              hipStream_t stream)
{
    (void)in_sizes; (void)n_in; (void)out_size; (void)ws_size;

    const float* x    = (const float*)d_in[0];
    const float* locs = (const float*)d_in[1];
    const float* fc0W = (const float*)d_in[2];
    const float* fc0b = (const float*)d_in[3];
    const float* fc1W = (const float*)d_in[4];
    const float* fc1b = (const float*)d_in[5];
    const float* lnS  = (const float*)d_in[6];
    const float* lnB  = (const float*)d_in[7];
    const float* Wq   = (const float*)d_in[8];
    const float* bq   = (const float*)d_in[9];
    const float* Wk   = (const float*)d_in[10];
    const float* bk   = (const float*)d_in[11];
    const float* Wv   = (const float*)d_in[12];
    const float* bv   = (const float*)d_in[13];
    const float* Wo   = (const float*)d_in[14];
    const float* bo   = (const float*)d_in[15];
    const float* mu   = (const float*)d_in[16];
    const float* sg   = (const float*)d_in[17];
    const float* Wg   = (const float*)d_in[18];
    const float* bg   = (const float*)d_in[19];

    float* out_x    = (float*)d_out;
    float* out_z    = out_x + (size_t)N_NODES * NCLASS;
    float* out_attn = out_z + (size_t)N_NODES * DIM;

    const size_t ND  = (size_t)N_NODES * DIM;       // 1M
    const size_t NHD = (size_t)N_NODES * HDIM;      // 8M
    const size_t NN  = (size_t)N_NODES * N_NODES;   // 4M

    char*  w   = (char*)d_ws;
    size_t off = 0;
    auto alloc = [&](size_t bytes) -> char* {
        char* p = w + off;
        off += (bytes + 255) & ~(size_t)255;
        return p;
    };
    float* z_f    = (float*)alloc(ND * 4);
    float* tmp    = (float*)alloc(ND * 4);
    bf16*  z_b    = (bf16*)alloc(ND * 2);
    bf16*  x_b    = (bf16*)alloc(ND * 2);
    bf16*  fc0T   = (bf16*)alloc((size_t)DIM * DIM * 2);
    bf16*  wqT    = (bf16*)alloc((size_t)DIM * HDIM * 2);
    bf16*  wkT    = (bf16*)alloc((size_t)DIM * HDIM * 2);
    bf16*  wvT    = (bf16*)alloc((size_t)DIM * HDIM * 2);
    bf16*  woT    = (bf16*)alloc((size_t)DIM * HDIM * 2);
    bf16*  q_b    = (bf16*)alloc(NHD * 2);
    bf16*  k_b    = (bf16*)alloc(NHD * 2);
    bf16*  vT_b   = (bf16*)alloc(NHD * 2);          // [H][D][N]
    bf16*  o_b    = (bf16*)alloc(NHD * 2);
    bf16*  attn_b = (bf16*)alloc(NN * HEADS * 2);

    const float inv_sqrt_d = 1.0f / sqrtf((float)DIM);
    const dim3  blk(256);
    const dim3  tblk(32, 8);

    // ---- input conversions + fc0 + LN/ELU ----
    cvt_bf16_kernel<<<(int)((ND + 255) / 256), blk, 0, stream>>>(x, x_b, (int)ND);
    transpose_bf16_kernel<<<dim3(DIM / 32, DIM / 32), tblk, 0, stream>>>(fc0W, fc0T, DIM, DIM);
    gemm_bf16_kernel<ST_F32><<<dim3(DIM / BN, N_NODES / BM, 1), blk, 0, stream>>>(
        x_b, DIM, 0, fc0T, DIM, 0, fc0b, tmp, nullptr, DIM, 0,
        N_NODES, DIM, DIM, 1.f, nullptr, nullptr, nullptr, nullptr, nullptr);
    ln_elu_kernel<<<N_NODES, blk, 0, stream>>>(tmp, nullptr, lnS, lnB, z_f, z_b);

    // ---- transformer layers ----
    for (int l = 0; l < LAYERS; ++l) {
        transpose_bf16_kernel<<<dim3(HDIM / 32, DIM / 32), tblk, 0, stream>>>(
            Wq + (size_t)l * DIM * HDIM, wqT, DIM, HDIM);
        transpose_bf16_kernel<<<dim3(HDIM / 32, DIM / 32), tblk, 0, stream>>>(
            Wk + (size_t)l * DIM * HDIM, wkT, DIM, HDIM);
        transpose_bf16_kernel<<<dim3(HDIM / 32, DIM / 32), tblk, 0, stream>>>(
            Wv + (size_t)l * DIM * HDIM, wvT, DIM, HDIM);
        transpose_bf16_kernel<<<dim3(DIM / 32, HDIM / 32), tblk, 0, stream>>>(
            Wo + (size_t)l * HDIM * DIM, woT, HDIM, DIM);

        dim3 gProj(HDIM / BN, N_NODES / BM, 1);
        gemm_bf16_kernel<ST_BF16><<<gProj, blk, 0, stream>>>(
            z_b, DIM, 0, wqT, DIM, 0, bq + (size_t)l * HDIM,
            nullptr, q_b, HDIM, 0, N_NODES, HDIM, DIM, 1.f,
            nullptr, nullptr, nullptr, nullptr, nullptr);
        gemm_bf16_kernel<ST_BF16><<<gProj, blk, 0, stream>>>(
            z_b, DIM, 0, wkT, DIM, 0, bk + (size_t)l * HDIM,
            nullptr, k_b, HDIM, 0, N_NODES, HDIM, DIM, 1.f,
            nullptr, nullptr, nullptr, nullptr, nullptr);
        gemm_bf16_kernel<ST_BF16_T><<<gProj, blk, 0, stream>>>(
            z_b, DIM, 0, wvT, DIM, 0, bv + (size_t)l * HDIM,
            nullptr, vT_b, N_NODES, 0, N_NODES, HDIM, DIM, 1.f,
            nullptr, nullptr, nullptr, nullptr, nullptr);

        // scores = Q Kᵀ * inv_sqrt_d + gaussian_bias -> f32 straight into d_out
        float* attnL = out_attn + (size_t)l * HEADS * NN;
        gemm_bf16_kernel<ST_F32 | EPI_SCORES>
            <<<dim3(N_NODES / BN, N_NODES / BM, HEADS), blk, 0, stream>>>(
            q_b, HDIM, DIM, k_b, HDIM, DIM, nullptr,
            attnL, nullptr, N_NODES, (long)NN,
            N_NODES, N_NODES, DIM, inv_sqrt_d,
            locs, mu + (size_t)l * 2, sg + (size_t)l * 2,
            Wg + (size_t)l * 16, bg + (size_t)l * 8);

        softmax_kernel<<<dim3(N_NODES, HEADS), blk, 0, stream>>>(attnL, attn_b);

        // o = attn @ V   (batched over heads; V already stored transposed)
        gemm_bf16_kernel<ST_BF16>
            <<<dim3(DIM / BN, N_NODES / BM, HEADS), blk, 0, stream>>>(
            attn_b, N_NODES, (long)NN, vT_b, N_NODES, (long)DIM * N_NODES, nullptr,
            nullptr, o_b, HDIM, DIM, N_NODES, DIM, N_NODES, 1.f,
            nullptr, nullptr, nullptr, nullptr, nullptr);

        // out projection
        gemm_bf16_kernel<ST_F32><<<dim3(DIM / BN, N_NODES / BM, 1), blk, 0, stream>>>(
            o_b, HDIM, 0, woT, HDIM, 0, bo + (size_t)l * DIM,
            tmp, nullptr, DIM, 0, N_NODES, DIM, HDIM, 1.f,
            nullptr, nullptr, nullptr, nullptr, nullptr);

        // residual + LN + ELU
        ln_elu_kernel<<<N_NODES, blk, 0, stream>>>(
            tmp, z_f, lnS + (size_t)(l + 1) * DIM, lnB + (size_t)(l + 1) * DIM, z_f, z_b);
    }

    // ---- outputs ----
    fc1_kernel<<<(N_NODES * NCLASS + 255) / 256, blk, 0, stream>>>(z_f, fc1W, fc1b, out_x);
    (void)hipMemcpyAsync(out_z, z_f, ND * sizeof(float), hipMemcpyDeviceToDevice, stream);
}
